// MWSA_6846177870250
// MI455X (gfx1250) — matrix operand order repaired and checked by reading, not compiled, not run on hardware
//
#include <hip/hip_runtime.h>

// MI455X / gfx1250, wave32. bf16 WMMA (v_wmma_f32_16x16x32_bf16) for all matmuls.
// Stage 1 (x3 branches): fused dwconv3x3 + shifted window attention -> bf16 y[pixel][160] in d_ws (80 MB).
// Stage 2: pointwise 96x144 GEMM via WMMA -> f32 out.

typedef __attribute__((ext_vector_type(16))) __bf16 v16bf;
typedef __attribute__((ext_vector_type(8)))  float  v8f;

__device__ __forceinline__ v8f zero8() {
  v8f z;
#pragma unroll
  for (int i = 0; i < 8; ++i) z[i] = 0.f;
  return z;
}

__device__ __forceinline__ v8f wmma_bf16(v16bf a, v16bf b, v8f c) {
  // (neg_a, A, neg_b, B, c_mod, C, reuse_a, reuse_b)
  return __builtin_amdgcn_wmma_f32_16x16x32_bf16(false, a, false, b, (short)0, c, false, false);
}

// A-matrix fragment (16x32 bf16): lane half h: elems 0..15 = row[32s+16h .. +15] (contiguous)
__device__ __forceinline__ v16bf load_a(const __bf16* rowp, int k0) {
  union { v16bf v; uint4 q[2]; } u;
  u.q[0] = *(const uint4*)(rowp + k0);
  u.q[1] = *(const uint4*)(rowp + k0 + 8);
  return u.v;
}
// B-matrix fragment (32x16 bf16, row-major-transposed storage [n][k]):
// lane half h: elems 0..15 = row[32s+16h .. +15] (contiguous, same k order as load_a)
__device__ __forceinline__ v16bf load_b(const __bf16* rowp, int k0) {
  union { v16bf v; uint4 q[2]; } u;
  u.q[0] = *(const uint4*)(rowp + k0);
  u.q[1] = *(const uint4*)(rowp + k0 + 8);
  return u.v;
}

// ---------------- Stage 1: fused dwconv + window attention ----------------
template<int WS, int IDX, int NW>
__global__ __launch_bounds__(NW*32)
void mwsa_stage1(const float* __restrict__ x, const float* __restrict__ w_dw,
                 __bf16* __restrict__ yws) {
  constexpr int P    = WS * WS;
  constexpr int Pk   = (P < 32) ? 32 : P;   // K-padded token count
  constexpr int T    = WS + 2;              // halo tile edge
  constexpr int SH   = WS / 2;              // shift
  constexpr int NWIN = 256 / WS;
  constexpr int NT   = P / 16;              // 16-wide column tiles of S
  constexpr int NTH  = NW * 32;
  constexpr int PADP = (Pk - P) > 0 ? (Pk - P) : 1;

  extern __shared__ __align__(16) char smem[];
  float*  xt = (float*)smem;                 // [32][T][T] input halo tile
  float*  wl = xt + 32 * T * T;              // [96][9] dw weights
  __bf16* Q  = (__bf16*)(wl + 96 * 9);       // [P][64]  (K 48..63 zero)
  __bf16* Vt = Q + P * 64;                   // [64][Pk] V transposed
  __bf16* Pl = Vt + 64 * Pk;                 // [NW][16][Pk] softmax rows per wave

  const int tid   = threadIdx.x;
  const int batch = blockIdx.x / (NWIN * NWIN);
  const int wrem  = blockIdx.x % (NWIN * NWIN);
  const int wi = wrem / NWIN, wj = wrem % NWIN;
  const int cin0 = IDX * 32;

  // ---- cooperative loads: x halo tile (circular wrap), weights, zero pads ----
  const int r0 = wi * WS + SH - 1, c0 = wj * WS + SH - 1;
  for (int t = tid; t < 32 * T * T; t += NTH) {
    int c = t / (T * T), rm = t % (T * T), a = rm / T, b2 = rm % T;
    int gr = (r0 + a) & 255, gc = (c0 + b2) & 255;
    xt[t] = x[(size_t)((batch * 96 + cin0 + c)) * 65536 + gr * 256 + gc];
  }
  for (int t = tid; t < 96 * 9; t += NTH) wl[t] = w_dw[IDX * 96 * 9 + t];
  for (int t = tid; t < P * 16; t += NTH) Q[(t >> 4) * 64 + 48 + (t & 15)] = (__bf16)0.f;
  if (Pk != P) {
    for (int t = tid; t < 64 * (Pk - P); t += NTH) {
      int ch = t / PADP, p = P + t % PADP; Vt[ch * Pk + p] = (__bf16)0.f;
    }
    for (int t = tid; t < NW * 16 * (Pk - P); t += NTH) {
      int rr = t / PADP, cc = P + t % PADP; Pl[rr * Pk + cc] = (__bf16)0.f;
    }
  }
  __syncthreads();

  // ---- depthwise conv 3x3 (zero-pad at true image borders, circular window) ----
  for (int t = tid; t < P * 96; t += NTH) {
    int p = t / 96, oc = t % 96;
    int pi = p / WS, pj = p % WS;
    int cin = oc / 3;  // local input channel
    int si = (wi * WS + SH + pi) & 255;
    int sj = (wj * WS + SH + pj) & 255;
    const float* wr = wl + oc * 9;
    const float* xb = xt + cin * T * T + pi * T + pj;
    float acc = 0.f;
#pragma unroll
    for (int di = -1; di <= 1; ++di) {
      bool rv = (unsigned)(si + di) < 256u;
#pragma unroll
      for (int dj = -1; dj <= 1; ++dj) {
        bool cv = (unsigned)(sj + dj) < 256u;
        float xv = (rv && cv) ? xb[(di + 1) * T + (dj + 1)] : 0.f;
        acc = fmaf(xv, wr[(di + 1) * 3 + (dj + 1)], acc);
      }
    }
    if (oc < 48) Q[p * 64 + oc] = (__bf16)acc;       // q half
    else         Vt[(oc - 48) * Pk + p] = (__bf16)acc; // v half, transposed
  }
  __syncthreads();

  // ---- attention: each wave owns 16 query rows per iteration ----
  const int wave = tid >> 5, lane = tid & 31;
  const int ln = lane & 15, hh = lane >> 4;
  __bf16* Pw = Pl + wave * 16 * Pk;

  for (int qb = wave; qb < NT; qb += NW) {  // uniform trip count across waves
    const int m0 = qb * 16;
    const __bf16* arow = Q + (m0 + ln) * 64;
    v16bf a0 = load_a(arow, 16 * hh);
    v16bf a1 = load_a(arow, 32 + 16 * hh);

    v8f acc[NT];
#pragma unroll
    for (int j = 0; j < NT; ++j) {
      const __bf16* br = Q + (j * 16 + ln) * 64;  // B = Q^T, column n = Q row n
      v8f c = zero8();
      c = wmma_bf16(a0, load_b(br, 16 * hh), c);
      c = wmma_bf16(a1, load_b(br, 32 + 16 * hh), c);
      acc[j] = c;
    }

    // softmax over each row (rows 0..7 live in lanes 0..15, rows 8..15 in lanes 16..31)
    float rinv[8];
#pragma unroll
    for (int r = 0; r < 8; ++r) {
      float mx = acc[0][r];
#pragma unroll
      for (int j = 1; j < NT; ++j) mx = fmaxf(mx, acc[j][r]);
#pragma unroll
      for (int msk = 1; msk < 16; msk <<= 1) mx = fmaxf(mx, __shfl_xor(mx, msk, 32));
      float s = 0.f;
#pragma unroll
      for (int j = 0; j < NT; ++j) { float e = __expf(acc[j][r] - mx); acc[j][r] = e; s += e; }
#pragma unroll
      for (int msk = 1; msk < 16; msk <<= 1) s += __shfl_xor(s, msk, 32);
      rinv[r] = 1.f / s;
    }
#pragma unroll
    for (int j = 0; j < NT; ++j)
#pragma unroll
      for (int r = 0; r < 8; ++r)
        Pw[(r + 8 * hh) * Pk + j * 16 + ln] = (__bf16)(acc[j][r] * rinv[r]);
    __syncthreads();

    // y = P @ V   (16 x 48)
    v8f y[3];
#pragma unroll
    for (int nt = 0; nt < 3; ++nt) y[nt] = zero8();
#pragma unroll
    for (int s = 0; s < Pk / 32; ++s) {
      v16bf ap = load_a(Pw + ln * Pk, 32 * s + 16 * hh);
#pragma unroll
      for (int nt = 0; nt < 3; ++nt) {
        v16bf bv = load_b(Vt + (nt * 16 + ln) * Pk, 32 * s + 16 * hh);
        y[nt] = wmma_bf16(ap, bv, y[nt]);
      }
    }
    // scatter to y_ws[pixel][160] at rolled-back (= source) coordinates, bf16
#pragma unroll
    for (int nt = 0; nt < 3; ++nt) {
      int ch = IDX * 48 + nt * 16 + ln;
#pragma unroll
      for (int r = 0; r < 8; ++r) {
        int p  = m0 + r + 8 * hh;
        int pi = p / WS, pj = p % WS;
        int si = (wi * WS + SH + pi) & 255;
        int sj = (wj * WS + SH + pj) & 255;
        yws[(size_t)(batch * 65536 + si * 256 + sj) * 160 + ch] = (__bf16)y[nt][r];
      }
    }
    __syncthreads();
  }
}

// ---------------- Stage 2: pointwise 96x144 GEMM over 262144 pixels ----------------
__global__ __launch_bounds__(256)
void mwsa_stage2(const __bf16* __restrict__ yws, const float* __restrict__ wout,
                 float* __restrict__ out) {
  __shared__ __align__(16) __bf16 wl[96 * 160];
  const int tid = threadIdx.x;
  for (int t = tid; t < 96 * 160; t += 256) {
    int o = t / 160, i = t % 160;
    wl[t] = (i < 144) ? (__bf16)wout[o * 144 + i] : (__bf16)0.f;
  }
  __syncthreads();

  const int wave = tid >> 5, lane = tid & 31, ln = lane & 15, hh = lane >> 4;
  const int pixbase = (blockIdx.x * 8 + wave) * 16;
  const int pix = pixbase + ln;
  const __bf16* yb = yws + (size_t)pix * 160;

  v16bf B[5];
#pragma unroll
  for (int s = 0; s < 5; ++s) {
    union { v16bf v; uint4 q[2]; } u;
    if (s == 4 && hh == 1) {           // K 144..159 is the zero pad
      u.q[0] = make_uint4(0u, 0u, 0u, 0u);
      u.q[1] = make_uint4(0u, 0u, 0u, 0u);
    } else {
      u.q[0] = *(const uint4*)(yb + 32 * s + 16 * hh);
      u.q[1] = *(const uint4*)(yb + 32 * s + 16 * hh + 8);
    }
    B[s] = u.v;
  }

  const int b  = pixbase >> 16;
  const int hw = (pixbase & 65535) + ln;
#pragma unroll
  for (int m = 0; m < 6; ++m) {
    v8f acc = zero8();
    const __bf16* ar = wl + (m * 16 + ln) * 160;
#pragma unroll
    for (int s = 0; s < 5; ++s)
      acc = wmma_bf16(load_a(ar, 32 * s + 16 * hh), B[s], acc);
#pragma unroll
    for (int r = 0; r < 8; ++r) {
      int o = m * 16 + r + 8 * hh;
      out[(size_t)(b * 96 + o) * 65536 + hw] = acc[r];
    }
  }
}

// ---------------- host ----------------
static constexpr size_t smem_stage1(int WS, int NW) {
  int P = WS * WS, Pk = (P < 32) ? 32 : P, T = WS + 2;
  return (size_t)4 * (32 * T * T + 96 * 9) + (size_t)2 * (P * 64 + 64 * Pk + NW * 16 * Pk);
}

extern "C" void kernel_launch(void* const* d_in, const int* in_sizes, int n_in,
                              void* d_out, int out_size, void* d_ws, size_t ws_size,
                              hipStream_t stream) {
  (void)in_sizes; (void)n_in; (void)out_size; (void)ws_size;
  const float* x     = (const float*)d_in[0];   // (4,96,256,256)
  const float* w_dw  = (const float*)d_in[1];   // (288,1,3,3)
  const float* w_out = (const float*)d_in[2];   // (96,144,1,1)
  float* out = (float*)d_out;                   // (4,96,256,256)
  __bf16* yws = (__bf16*)d_ws;                  // [262144][160] bf16 = 80 MB

  // ws=4 : 16384 windows, 1 wave each; ws=8 : 4096 windows, 4 waves; ws=16 : 1024 windows, 8 waves
  mwsa_stage1<4, 0, 1><<<dim3(4 * 64 * 64), dim3(32),  smem_stage1(4, 1),  stream>>>(x, w_dw, yws);
  mwsa_stage1<8, 1, 4><<<dim3(4 * 32 * 32), dim3(128), smem_stage1(8, 4),  stream>>>(x, w_dw, yws);
  mwsa_stage1<16, 2, 8><<<dim3(4 * 16 * 16), dim3(256), smem_stage1(16, 8), stream>>>(x, w_dw, yws);
  // 262144 pixels / (8 waves * 16 px) = 2048 workgroups
  mwsa_stage2<<<dim3(2048), dim3(256), 0, stream>>>(yws, w_out, out);
}
